// EncoderLayer_58566174049067
// MI455X (gfx1250) — matrix operand order repaired and checked by reading, not compiled, not run on hardware
//
#include <hip/hip_runtime.h>
#include <hip/hip_bf16.h>
#include <math.h>

typedef __attribute__((ext_vector_type(16))) __bf16 v16bf;
typedef __attribute__((ext_vector_type(8)))  __bf16 v8bf;
typedef __attribute__((ext_vector_type(8)))  float  v8f;

#define NB   8
#define LSEQ 2048
#define EDIM 1024
#define HDIM 1024
#define NLROWS (NB * LSEQ)   // 16384

// ---------------------------------------------------------------------------
// pack two contiguous 8x-bf16 chunks into the 16-element WMMA operand vector
// ---------------------------------------------------------------------------
__device__ __forceinline__ v16bf pack16(const __bf16* p0, const __bf16* p1) {
    v8bf a = *(const v8bf*)p0;
    v8bf b = *(const v8bf*)p1;
    v16bf r;
#pragma unroll
    for (int i = 0; i < 8; ++i) { r[i] = a[i]; r[i + 8] = b[i]; }
    return r;
}

// ---------------------------------------------------------------------------
// float32 -> bf16 convert (straight copy)
// ---------------------------------------------------------------------------
__global__ __launch_bounds__(256) void cvt_bf16_kernel(
    const float* __restrict__ src, __bf16* __restrict__ dst, long n) {
    long i = (long)blockIdx.x * blockDim.x + threadIdx.x;
    if (i < n) dst[i] = (__bf16)src[i];
}

// ---------------------------------------------------------------------------
// float32 (rows x cols) -> bf16 transposed (cols x rows): weights become
// N-major x K so WMMA B-fragments are contiguous 16B loads per lane.
// ---------------------------------------------------------------------------
__global__ __launch_bounds__(256) void cvt_transpose_kernel(
    const float* __restrict__ src, __bf16* __restrict__ dst, int rows, int cols) {
    long i = (long)blockIdx.x * blockDim.x + threadIdx.x;
    if (i >= (long)rows * cols) return;
    int r = (int)(i / cols);
    int c = (int)(i % cols);
    dst[(long)c * rows + r] = (__bf16)src[i];
}

// ---------------------------------------------------------------------------
// WMMA GEMM: C = (A[MxK,bf16] * Bt[NxK,bf16]^T + bias) * scale (+ residual)
// One wave = 64x32 register macro-tile (4 M-tiles x 2 N-tiles):
//   per k-step 6 fragment loads feed 8 WMMAs (1.5 loads/WMMA).
// Block = 8 waves (2 M x 4 N) covering 128x128.
// A-frag: lane holds row (lane&15), halves of K per ISA 16-bit A layout.
// B-frag: lane holds col (lane&15), K split as for A: 8*(lane>>4)+0..7 and 16+8*(lane>>4)+0..7.
// C-frag: lane holds col (lane&15), rows r + 8*(lane>>4), r = 0..7.
// ---------------------------------------------------------------------------
__global__ __launch_bounds__(256) void wmma_gemm_kernel(
    const __bf16* __restrict__ A,     // M x K row-major
    const __bf16* __restrict__ Bt,    // N x K row-major (pre-transposed B)
    const float*  __restrict__ bias,  // [N] or null
    const float*  __restrict__ resid, // M x N f32 or null
    float scale,
    float*  __restrict__ outF,        // M x N f32 or null
    __bf16* __restrict__ outB,        // M x N bf16 or null
    int M, int N, int K)
{
    const int lane = threadIdx.x & 31;
    const int wave = threadIdx.x >> 5;
    const int hi   = lane >> 4;       // half-wave select
    const int r16  = lane & 15;

    const int mBase = blockIdx.x * 128 + (wave & 1) * 64;   // 4 M-tiles of 16
    const int nBase = blockIdx.y * 128 + (wave >> 1) * 32;  // 2 N-tiles of 16
    if (mBase >= M || nBase >= N) return;

    const __bf16* aRow[4];
    const __bf16* bRow[2];
#pragma unroll
    for (int i = 0; i < 4; ++i) aRow[i] = A  + (long)(mBase + 16 * i + r16) * K;
#pragma unroll
    for (int j = 0; j < 2; ++j) bRow[j] = Bt + (long)(nBase + 16 * j + r16) * K;

    v8f acc[4][2] = {};
    for (int k0 = 0; k0 < K; k0 += 32) {
        __builtin_prefetch(aRow[0] + k0 + 256, 0, 0);   // global_prefetch_b8
        v16bf af[4], bf[2];
#pragma unroll
        for (int i = 0; i < 4; ++i)
            af[i] = pack16(aRow[i] + k0 + hi * 8, aRow[i] + k0 + 16 + hi * 8);
#pragma unroll
        for (int j = 0; j < 2; ++j)
            bf[j] = pack16(bRow[j] + k0 + hi * 8, bRow[j] + k0 + 16 + hi * 8);
#pragma unroll
        for (int i = 0; i < 4; ++i)
#pragma unroll
            for (int j = 0; j < 2; ++j)
                acc[i][j] = __builtin_amdgcn_wmma_f32_16x16x32_bf16(
                    false, af[i], false, bf[j], (short)0, acc[i][j], false, false);
    }

#pragma unroll
    for (int j = 0; j < 2; ++j) {
        const int n = nBase + 16 * j + r16;
        const float bval = bias ? bias[n] : 0.0f;
#pragma unroll
        for (int i = 0; i < 4; ++i) {
#pragma unroll
            for (int r = 0; r < 8; ++r) {
                const int m = mBase + 16 * i + r + hi * 8;
                float v = (acc[i][j][r] + bval) * scale;
                if (resid) v += resid[(long)m * N + n];
                if (outF) outF[(long)m * N + n] = v;
                if (outB) outB[(long)m * N + n] = (__bf16)v;
            }
        }
    }
}

// ---------------------------------------------------------------------------
// Attention diagonal-softmax: for each key column b, stream all query tiles a,
// S[a,b] = Q[a,:].K[b,:] via WMMA; online max/sum over a; capture S[b,b].
// diag[n,b] = exp(S[b,b]-max)/sum. One wave per 16-wide key tile.
// The key-row B fragment is invariant in a, so 4 query tiles share it per
// k-sweep (10 loads / 4 WMMAs). Q,K (bf16, 64MB) stay L2-resident (192MB).
// ---------------------------------------------------------------------------
__global__ __launch_bounds__(256) void attn_diag_kernel(
    const __bf16* __restrict__ Qb,   // (N*L) x H bf16 (pre-scaled)
    const __bf16* __restrict__ Kb,   // (N*L) x H bf16 (pre-scaled)
    float* __restrict__ diag,        // N*L
    int L, int H)
{
    const int lane  = threadIdx.x & 31;
    const int wave  = threadIdx.x >> 5;
    const int hi    = lane >> 4;
    const int r16   = lane & 15;
    const int lt    = L >> 4;                       // key tiles per batch
    const int bTile = blockIdx.x * 8 + wave;        // global key-tile id
    const int n     = bTile / lt;
    const int b0    = (bTile % lt) * 16;
    const int bCol  = b0 + r16;                     // this lane's key index

    const __bf16* Qn   = Qb + (long)n * L * H;
    const __bf16* kRow = Kb + (long)n * L * H + (long)bCol * H;

    float runMax = -INFINITY, runSum = 0.0f, diagRaw = -INFINITY;

    for (int a0 = 0; a0 < L; a0 += 64) {            // 4 query tiles per sweep
        const __bf16* qRow[4];
#pragma unroll
        for (int i = 0; i < 4; ++i) qRow[i] = Qn + (long)(a0 + 16 * i + r16) * H;

        v8f acc[4] = {};
        for (int k0 = 0; k0 < H; k0 += 32) {
            __builtin_prefetch(qRow[0] + k0 + 256, 0, 0);
            const v16bf bv = pack16(kRow + k0 + hi * 8, kRow + k0 + 16 + hi * 8);
#pragma unroll
            for (int i = 0; i < 4; ++i) {
                const v16bf av = pack16(qRow[i] + k0 + hi * 8,
                                        qRow[i] + k0 + 16 + hi * 8);
                acc[i] = __builtin_amdgcn_wmma_f32_16x16x32_bf16(
                    false, av, false, bv, (short)0, acc[i], false, false);
            }
        }
        // online softmax over the 4 sub-tiles; lane holds rows r+8*hi of each
#pragma unroll
        for (int i = 0; i < 4; ++i) {
            float tmax = acc[i][0];
#pragma unroll
            for (int r = 1; r < 8; ++r) tmax = fmaxf(tmax, acc[i][r]);
            const float newMax = fmaxf(runMax, tmax);
            float s = 0.0f;
#pragma unroll
            for (int r = 0; r < 8; ++r) s += __expf(acc[i][r] - newMax);
            runSum = runSum * __expf(runMax - newMax) + s;
            runMax = newMax;
#pragma unroll
            for (int r = 0; r < 8; ++r)
                if (a0 + 16 * i + r + hi * 8 == bCol) diagRaw = acc[i][r];
        }
    }

    // column stats are split across lane pair (l, l^16): merge
    const float oM = __shfl_xor(runMax,  16, 32);
    const float oS = __shfl_xor(runSum,  16, 32);
    const float oD = __shfl_xor(diagRaw, 16, 32);
    const float M  = fmaxf(runMax, oM);
    const float S  = runSum * __expf(runMax - M) + oS * __expf(oM - M);
    const float D  = fmaxf(diagRaw, oD);            // non-owner holds -inf

    if (hi == 0) diag[(long)n * L + bCol] = __expf(D - M) / S;
}

// ---------------------------------------------------------------------------
// Fused O = diag*V + X, then LayerNorm(g1,b1). One block (256 thr) per row.
// Writes f32 (for FF residual) and bf16 (FF GEMM operand).
// ---------------------------------------------------------------------------
__global__ __launch_bounds__(256) void fuse_ln1_kernel(
    const float* __restrict__ V, const float* __restrict__ X,
    const float* __restrict__ diag,
    const float* __restrict__ g1, const float* __restrict__ b1,
    float* __restrict__ h1, __bf16* __restrict__ h1b, int H)
{
    const long row = blockIdx.x;
    const float d  = diag[row];
    const float* Vr = V + row * H;
    const float* Xr = X + row * H;

    float vals[4], s = 0.0f, s2 = 0.0f;
#pragma unroll
    for (int i = 0; i < 4; ++i) {
        const int idx = threadIdx.x + i * 256;
        const float y = d * Vr[idx] + Xr[idx];
        vals[i] = y; s += y; s2 += y * y;
    }
    for (int off = 16; off; off >>= 1) {
        s  += __shfl_xor(s,  off, 32);
        s2 += __shfl_xor(s2, off, 32);
    }
    __shared__ float rs[8], rs2[8];
    const int wave = threadIdx.x >> 5, lane = threadIdx.x & 31;
    if (lane == 0) { rs[wave] = s; rs2[wave] = s2; }
    __syncthreads();
    if (wave == 0) {
        float a  = (lane < 8) ? rs[lane]  : 0.0f;
        float a2 = (lane < 8) ? rs2[lane] : 0.0f;
        for (int off = 4; off; off >>= 1) {
            a  += __shfl_xor(a,  off, 32);
            a2 += __shfl_xor(a2, off, 32);
        }
        if (lane == 0) { rs[0] = a; rs2[0] = a2; }
    }
    __syncthreads();
    const float mean = rs[0] / H;
    const float var  = rs2[0] / H - mean * mean;
    const float inv  = rsqrtf(var + 1e-5f);
#pragma unroll
    for (int i = 0; i < 4; ++i) {
        const int idx = threadIdx.x + i * 256;
        const float yn = (vals[i] - mean) * inv * g1[idx] + b1[idx];
        h1[row * H + idx]  = yn;
        h1b[row * H + idx] = (__bf16)yn;
    }
}

// ---------------------------------------------------------------------------
// Final LayerNorm(g2,b2) over the FF output. One block per row.
// ---------------------------------------------------------------------------
__global__ __launch_bounds__(256) void ln2_kernel(
    const float* __restrict__ Y,
    const float* __restrict__ g2, const float* __restrict__ b2,
    float* __restrict__ out, int H)
{
    const long row = blockIdx.x;
    const float* Yr = Y + row * H;
    float vals[4], s = 0.0f, s2 = 0.0f;
#pragma unroll
    for (int i = 0; i < 4; ++i) {
        const int idx = threadIdx.x + i * 256;
        const float y = Yr[idx];
        vals[i] = y; s += y; s2 += y * y;
    }
    for (int off = 16; off; off >>= 1) {
        s  += __shfl_xor(s,  off, 32);
        s2 += __shfl_xor(s2, off, 32);
    }
    __shared__ float rs[8], rs2[8];
    const int wave = threadIdx.x >> 5, lane = threadIdx.x & 31;
    if (lane == 0) { rs[wave] = s; rs2[wave] = s2; }
    __syncthreads();
    if (wave == 0) {
        float a  = (lane < 8) ? rs[lane]  : 0.0f;
        float a2 = (lane < 8) ? rs2[lane] : 0.0f;
        for (int off = 4; off; off >>= 1) {
            a  += __shfl_xor(a,  off, 32);
            a2 += __shfl_xor(a2, off, 32);
        }
        if (lane == 0) { rs[0] = a; rs2[0] = a2; }
    }
    __syncthreads();
    const float mean = rs[0] / H;
    const float var  = rs2[0] / H - mean * mean;
    const float inv  = rsqrtf(var + 1e-5f);
#pragma unroll
    for (int i = 0; i < 4; ++i) {
        const int idx = threadIdx.x + i * 256;
        out[row * H + idx] = (vals[i] - mean) * inv * g2[idx] + b2[idx];
    }
}

// ---------------------------------------------------------------------------
extern "C" void kernel_launch(void* const* d_in, const int* in_sizes, int n_in,
                              void* d_out, int out_size, void* d_ws, size_t ws_size,
                              hipStream_t stream) {
    (void)in_sizes; (void)n_in; (void)out_size; (void)ws_size;
    const float* X  = (const float*)d_in[0];
    const float* Wq = (const float*)d_in[1];
    const float* bq = (const float*)d_in[2];
    const float* Wk = (const float*)d_in[3];
    const float* bk = (const float*)d_in[4];
    const float* Wv = (const float*)d_in[5];
    const float* bv = (const float*)d_in[6];
    const float* Wf = (const float*)d_in[7];
    const float* bf = (const float*)d_in[8];
    const float* g1 = (const float*)d_in[9];
    const float* b1 = (const float*)d_in[10];
    const float* g2 = (const float*)d_in[11];
    const float* b2 = (const float*)d_in[12];
    float* out = (float*)d_out;

    // ---- workspace layout (regions reused across phases) ----
    char* w = (char*)d_ws;
    size_t off = 0;
    auto take = [&](size_t bytes) -> char* {
        char* p = w + off;
        off = (off + bytes + 255) & ~(size_t)255;
        return p;
    };
    const size_t wB  = (size_t)EDIM * HDIM * sizeof(__bf16);          // 2 MB
    const size_t xB  = (size_t)NLROWS * EDIM * sizeof(__bf16);        // 32 MB
    const size_t fB  = (size_t)NLROWS * HDIM * sizeof(float);         // 64 MB

    __bf16* Wqt = (__bf16*)take(wB);
    __bf16* Wkt = (__bf16*)take(wB);
    __bf16* Wvt = (__bf16*)take(wB);
    __bf16* Wft = (__bf16*)take(wB);
    char*   xRegion  = take(xB);          // Xb (phase 1-2) -> h1b (phase 4-5)
    char*   qkRegion = take(2 * xB);      // Qb+Kb (2-3)    -> FF output f32 (5-6)
    float*  Vf   = (float*)take(fB);
    float*  diag = (float*)take((size_t)NLROWS * sizeof(float));
    float*  h1   = (float*)take(fB);

    __bf16* Xb  = (__bf16*)xRegion;
    __bf16* h1b = (__bf16*)xRegion;
    __bf16* Qb  = (__bf16*)qkRegion;
    __bf16* Kb  = (__bf16*)(qkRegion + xB);
    float*  yFF = (float*)qkRegion;

    const dim3 blk(256);
    const float scaleQK = 0.03125f;  // 1/sqrt(1024)

    // 1) convert + transpose weights; convert X
    cvt_transpose_kernel<<<dim3((EDIM * HDIM) / 256), blk, 0, stream>>>(Wq, Wqt, EDIM, HDIM);
    cvt_transpose_kernel<<<dim3((EDIM * HDIM) / 256), blk, 0, stream>>>(Wk, Wkt, EDIM, HDIM);
    cvt_transpose_kernel<<<dim3((EDIM * HDIM) / 256), blk, 0, stream>>>(Wv, Wvt, EDIM, HDIM);
    cvt_transpose_kernel<<<dim3((HDIM * HDIM) / 256), blk, 0, stream>>>(Wf, Wft, HDIM, HDIM);
    cvt_bf16_kernel<<<dim3((NLROWS * EDIM) / 256), blk, 0, stream>>>(X, Xb, (long)NLROWS * EDIM);

    // 2) Q/K/V projections via WMMA (block tile 128x128)
    const dim3 gGemm(NLROWS / 128, HDIM / 128);
    wmma_gemm_kernel<<<gGemm, blk, 0, stream>>>(Xb, Wqt, bq, nullptr, scaleQK,
                                                nullptr, Qb, NLROWS, HDIM, EDIM);
    wmma_gemm_kernel<<<gGemm, blk, 0, stream>>>(Xb, Wkt, bk, nullptr, scaleQK,
                                                nullptr, Kb, NLROWS, HDIM, EDIM);
    wmma_gemm_kernel<<<gGemm, blk, 0, stream>>>(Xb, Wvt, bv, nullptr, 1.0f,
                                                Vf, nullptr, NLROWS, HDIM, EDIM);

    // 3) diagonal of query-axis softmax (online, no LxL buffer)
    attn_diag_kernel<<<dim3(NLROWS / 16 / 8), blk, 0, stream>>>(Qb, Kb, diag, LSEQ, HDIM);

    // 4) O = diag*V + X, LayerNorm1
    fuse_ln1_kernel<<<dim3(NLROWS), blk, 0, stream>>>(Vf, X, diag, g1, b1, h1, h1b, HDIM);

    // 5) FF GEMM + bias + residual (WMMA)
    wmma_gemm_kernel<<<gGemm, blk, 0, stream>>>(h1b, Wft, bf, h1, 1.0f,
                                                yFF, nullptr, NLROWS, HDIM, HDIM);

    // 6) LayerNorm2 -> output
    ln2_kernel<<<dim3(NLROWS), blk, 0, stream>>>(yFF, g2, b2, out, HDIM);
}
